// BCHConv2D_58171037057636
// MI455X (gfx1250) — matrix operand order repaired and checked by reading, not compiled, not run on hardware
//
#include <hip/hip_runtime.h>

typedef __attribute__((ext_vector_type(16))) _Float16 v16h;
typedef __attribute__((ext_vector_type(8)))  _Float16 v8h;
typedef __attribute__((ext_vector_type(4)))  _Float16 v4h;
typedef __attribute__((ext_vector_type(8)))  float    v8f;
typedef __attribute__((ext_vector_type(4)))  float    v4f;

#define KH 7
#define KW 7
#define CIN 32
#define STREAMS 16
#define MDEG 6
#define NRP 4
#define OC 112          // STREAMS*(MDEG+1)
#define OC2 224
#define NTILES 14       // OC2/16
#define HP 122
#define WP 122
#define OUTC 256
#define BTHREADS 224    // 7 waves
#define MT 32           // pixels per block (2 WMMA M-tiles)
#define SACOLS 38       // MT + KW - 1 staged x columns
#define ROWSTRIDE 40    // halfs per sA row (padded: 80B stride, conflict-free)
#define BFRAG 512       // halfs per (tap, ntile) B fragment: 32 lanes * 16

__device__ __forceinline__ float tri_f(float v) {
    float a = fabsf(v);
    if (a > 1.0f) return 0.0f;
    return (v < 0.0f) ? (v + 1.0f) : (1.0f - v);
}

// ---------------------------------------------------------------------------
// Kernel 1: bake harmonic filter into WMMA B-fragment layout (f16).
// filtB[((tap*14 + nt)*32 + lane)*16 + j], tap = ky*7+kx.
// lane<16: col N = nt*16+lane, K = j; lane>=16: col N = nt*16+lane-16, K=16+j.
// ---------------------------------------------------------------------------
__global__ void build_filter_kernel(const float* __restrict__ w,
                                    _Float16* __restrict__ filtB) {
    int blk = blockIdx.x;             // 0 .. 49*14-1
    int tap = blk / NTILES;
    int nt  = blk % NTILES;
    int ky = tap / KW, kx = tap % KW;
    int lane = threadIdx.x;
    int col = nt * 16 + (lane & 15);
    bool isImag = (col >= OC);
    int o = isImag ? (col - OC) : col;
    int s = o / (MDEG + 1);
    int n = o % (MDEG + 1);

    float norm2[NRP] = {0.f, 0.f, 0.f, 0.f};
    for (int hy = 0; hy < KH; ++hy) {
        for (int hx = 0; hx < KW; ++hx) {
            float gx = (float)(hx - 3), gy = (float)(hy - 3);
            float rad = sqrtf(gx * gx + gy * gy);
            bool center = (hy == 3 && hx == 3);
            for (int rp = 0; rp < NRP; ++rp) {
                float t = tri_f(rad - (float)rp);
                if (center && n >= 1) t = 0.f;
                norm2[rp] += t * t;
            }
        }
    }
    float gx = (float)(kx - 3), gy = (float)(ky - 3);
    float rad = sqrtf(gx * gx + gy * gy);
    float th = atan2f(gy, gx);
    float cn = cosf((float)n * th);
    float sn = sinf((float)n * th);
    bool center = (ky == 3 && kx == 3);
    float acomp[NRP];
    for (int rp = 0; rp < NRP; ++rp) {
        float t = tri_f(rad - (float)rp);
        if (center && n >= 1) t = 0.f;
        float nr = sqrtf(norm2[rp]);
        if (nr == 0.f) nr = 1.f;
        float re = t * cn / nr;
        float im = -t * sn / nr;
        acomp[rp] = isImag ? im : re;
    }
    _Float16* dst = filtB + ((size_t)blk * 32 + lane) * 16;
    for (int j = 0; j < 16; ++j) {
        int c = (lane < 16) ? j : (16 + j);
        const float* wp = w + (((size_t)c * STREAMS + s) * (MDEG + 1) + n) * NRP;
        float v = acomp[0] * wp[0] + acomp[1] * wp[1] +
                  acomp[2] * wp[2] + acomp[3] * wp[3];
        dst[j] = (_Float16)v;
    }
}

// ---------------------------------------------------------------------------
// Kernel 2: fused implicit-GEMM conv (f16 WMMA, f32 acc) + bispectrum.
// Block: 32 pixels x 224 conv channels. 7 waves x (2 Mtiles x 2 Ntiles).
// Per ky: stage one 38x32 x-slab once; 7 kx taps read shifted LDS windows.
// ---------------------------------------------------------------------------
__global__ void __launch_bounds__(BTHREADS)
conv_bisp_kernel(const float* __restrict__ x,
                 const _Float16* __restrict__ filtB,
                 const float* __restrict__ bias,
                 float* __restrict__ out) {
    const int xt = blockIdx.x;        // 0..3 -> x0 = 32*xt
    const int y  = blockIdx.y;        // 0..121
    const int b  = blockIdx.z;        // 0..7
    const int x0 = xt * MT;
    const int tid  = threadIdx.x;
    const int lane = tid & 31;
    const int wave = tid >> 5;        // 0..6
    const int nt0 = wave * 2;
    const int nt1 = wave * 2 + 1;

    __shared__ __align__(16) _Float16 sA[SACOLS * ROWSTRIDE];
    __shared__ float sFM[MT * OC2];

    v8f acc00 = {0.f,0.f,0.f,0.f,0.f,0.f,0.f,0.f};
    v8f acc01 = acc00, acc10 = acc00, acc11 = acc00;

    // Fixed per-thread staging chunks (float4 of channels): 304 chunks total.
    const int i0 = tid, i1 = tid + BTHREADS;
    const int j0 = i0 >> 3, c0 = (i0 & 7) * 4;
    const int j1 = i1 >> 3, c1 = (i1 & 7) * 4;
    int col0 = x0 + j0; if (col0 > 127) col0 = 127;   // clamp; masked at store
    int col1 = x0 + j1; if (col1 > 127) col1 = 127;
    const bool has1 = (i1 < SACOLS * 8);
    const float* xrow = x + (((size_t)b * 128 + y) * 128) * CIN;
    const float* p0 = xrow + (size_t)col0 * CIN + c0;
    const float* p1 = xrow + (size_t)col1 * CIN + c1;
    _Float16* s0 = sA + j0 * ROWSTRIDE + c0;
    _Float16* s1 = sA + j1 * ROWSTRIDE + c1;

    // A-fragment LDS base (contiguous K: lane half h holds K 16h..16h+15)
    const _Float16* aBase = sA + (lane & 15) * ROWSTRIDE + (lane >> 4) * 16;
    // B-fragment base per lane
    const _Float16* bBase = filtB + lane * 16;

    for (int ky = 0; ky < KH; ++ky) {
        {   // stage 38 x 32 slab for row y+ky
            v4f v = *(const v4f*)p0;
            v4h h; h[0] = (_Float16)v[0]; h[1] = (_Float16)v[1];
                   h[2] = (_Float16)v[2]; h[3] = (_Float16)v[3];
            *(v4h*)s0 = h;
            if (has1) {
                v4f u = *(const v4f*)p1;
                v4h g; g[0] = (_Float16)u[0]; g[1] = (_Float16)u[1];
                       g[2] = (_Float16)u[2]; g[3] = (_Float16)u[3];
                *(v4h*)s1 = g;
            }
        }
        __syncthreads();

        const _Float16* bky = bBase + (size_t)ky * KW * NTILES * BFRAG;
        #pragma unroll
        for (int kx = 0; kx < KW; ++kx) {
            v8h a0lo = *(const v8h*)(aBase + kx * ROWSTRIDE);
            v8h a0hi = *(const v8h*)(aBase + kx * ROWSTRIDE + 8);
            v16h a0 = __builtin_shufflevector(a0lo, a0hi,
                        0,1,2,3,4,5,6,7,8,9,10,11,12,13,14,15);
            v8h a1lo = *(const v8h*)(aBase + (16 + kx) * ROWSTRIDE);
            v8h a1hi = *(const v8h*)(aBase + (16 + kx) * ROWSTRIDE + 8);
            v16h a1 = __builtin_shufflevector(a1lo, a1hi,
                        0,1,2,3,4,5,6,7,8,9,10,11,12,13,14,15);

            v16h bf0 = *(const v16h*)(bky + (kx * NTILES + nt0) * BFRAG);
            v16h bf1 = *(const v16h*)(bky + (kx * NTILES + nt1) * BFRAG);

            acc00 = __builtin_amdgcn_wmma_f32_16x16x32_f16(
                        false, a0, false, bf0, (short)0, acc00, false, false);
            acc01 = __builtin_amdgcn_wmma_f32_16x16x32_f16(
                        false, a0, false, bf1, (short)0, acc01, false, false);
            acc10 = __builtin_amdgcn_wmma_f32_16x16x32_f16(
                        false, a1, false, bf0, (short)0, acc10, false, false);
            acc11 = __builtin_amdgcn_wmma_f32_16x16x32_f16(
                        false, a1, false, bf1, (short)0, acc11, false, false);
        }
        __syncthreads();          // protect sA before next ky restage
        p0 += 128 * CIN;          // next input row
        p1 += 128 * CIN;
    }

    // Scatter C fragments (lane = N, vgpr = M, high-half lanes at M+8)
    {
        int mofs = (lane >> 4) * 8;
        int ncol = lane & 15;
        #pragma unroll
        for (int v = 0; v < 8; ++v) {
            int m = v + mofs;
            sFM[m * OC2 + nt0 * 16 + ncol]        = acc00[v];
            sFM[m * OC2 + nt1 * 16 + ncol]        = acc01[v];
            sFM[(m + 16) * OC2 + nt0 * 16 + ncol] = acc10[v];
            sFM[(m + 16) * OC2 + nt1 * 16 + ncol] = acc11[v];
        }
    }
    __syncthreads();

    // Bispectrum + bias + ReLU: 32 pixels x 256 channels, 4 channels/thread.
    // Nibble tables over slot (channel>>4): n1/n2/imag-flag.
    const unsigned long long T1 = 0x3322221111110000ull;
    const unsigned long long T2 = 0x3333223322113210ull;
    const unsigned long long TI = 0x1010101010100000ull;
    for (int idx = tid; idx < MT * (OUTC / 4); idx += BTHREADS) {
        int m  = idx >> 6;                 // pixel 0..31
        int ch = (idx & 63) * 4;           // channel 0,4,...,252
        int px = x0 + m;
        if (px > WP - 1) continue;
        int slot = ch >> 4;
        int sh = slot * 4;
        int n1 = (int)((T1 >> sh) & 7);
        int n2 = (int)((T2 >> sh) & 7);
        int isim = (int)((TI >> sh) & 1);
        int n3 = n1 + n2;
        const float* fmm = sFM + m * OC2;
        v4f r;
        #pragma unroll
        for (int q = 0; q < 4; ++q) {
            int s = (ch & 15) + q;
            float ar = fmm[s * 7 + n1], ai = fmm[OC + s * 7 + n1];
            float br = fmm[s * 7 + n2], bi = fmm[OC + s * 7 + n2];
            float er = fmm[s * 7 + n3], ei = fmm[OC + s * 7 + n3];
            float pr = ar * br - ai * bi;
            float pi = ar * bi + ai * br;
            float re = pr * er + pi * ei;
            float im = pi * er - pr * ei;
            r[q] = isim ? im : re;
        }
        v4f bb = *(const v4f*)(bias + ch);
        r += bb;
        #pragma unroll
        for (int q = 0; q < 4; ++q) r[q] = fmaxf(r[q], 0.f);
        *(v4f*)(out + (((size_t)b * HP + y) * WP + px) * OUTC + ch) = r;
    }
}

extern "C" void kernel_launch(void* const* d_in, const int* in_sizes, int n_in,
                              void* d_out, int out_size, void* d_ws, size_t ws_size,
                              hipStream_t stream) {
    (void)in_sizes; (void)n_in; (void)out_size; (void)ws_size;
    const float* x    = (const float*)d_in[0];
    const float* w    = (const float*)d_in[1];
    const float* bias = (const float*)d_in[2];
    float* out = (float*)d_out;
    _Float16* filtB = (_Float16*)d_ws;   // 49*14*512 f16 = ~686 KB

    build_filter_kernel<<<dim3(KH * KW * NTILES), dim3(32), 0, stream>>>(w, filtB);
    conv_bisp_kernel<<<dim3(4, HP, 8), dim3(BTHREADS), 0, stream>>>(x, filtB, bias, out);
}
